// Discretized_GRU_45097156608137
// MI455X (gfx1250) — matrix operand order repaired and checked by reading, not compiled, not run on hardware
//
#include <hip/hip_runtime.h>
#include <hip/hip_bf16.h>

typedef __attribute__((ext_vector_type(16))) _Float16 v16h;
typedef __attribute__((ext_vector_type(8)))  _Float16 v8h;
typedef __attribute__((ext_vector_type(8)))  float    v8f;

#define S_   48
#define O_   1024
#define B_   2048
#define D_   64
#define H_   512
#define P_   10
#define PH_  512
#define COV_  32
#define COVH_ 64
#define CLSH_ 32
#define LOGC_ 0.9189385332046727f   /* log(sqrt(2*pi)) */
#define OBSSTD_ 0.01f

// ---------------------------------------------------------------------------
// WMMA GEMM:  C[M,N] = act( A[M,K](f16) * W^T + bias[N] )
// A: f16 [M,K] row-major.  Wp: fragment-packed f16 weights (see pack_b_kernel):
//   Wp[((ntile*(K/32) + kchunk)*32 + lane)*16 + j]
//   where  n = ntile*16 + lane%16,  k = kchunk*32 + (j/8)*16 + (lane/16)*8 + j%8
// Each wave computes a 32x64 tile: 2 A-fragments x 4 B-fragments = 8 WMMAs
// per 32-deep k-chunk; every B fragment is one coalesced 1KB block load.
// Block = 256 thr = 8 waves stacked in M (256 rows); grid = (M/256, N/64).
// Optional dual store: f32 C and/or f16 C16.
// ---------------------------------------------------------------------------
__global__ __launch_bounds__(256) void wmma_gemm_kernel(
    const _Float16* __restrict__ A, int lda,
    const _Float16* __restrict__ Wp,
    const float* __restrict__ bias,           // may be null
    float* __restrict__ C,                    // may be null
    _Float16* __restrict__ C16,               // may be null
    int ldc, int K, int do_relu)
{
  const int lane = threadIdx.x & 31;
  const int wave = threadIdx.x >> 5;
  const int m0   = blockIdx.x * 256 + wave * 32;
  const int n0   = blockIdx.y * 64;
  const int hi   = lane >> 4;
  const int col  = lane & 15;
  const int rowA = m0 + col;
  const int rowB = m0 + 16 + col;
  const int kchunks = K >> 5;

  v8f acc[2][4] = {};

  for (int kc = 0; kc < kchunks; ++kc) {
    const int k0 = kc << 5;
    const _Float16* a0 = A + (size_t)rowA * lda + k0 + hi * 8;
    const _Float16* a1 = A + (size_t)rowB * lda + k0 + hi * 8;
    v8h lo0 = *(const v8h*)a0, up0 = *(const v8h*)(a0 + 16);
    v8h lo1 = *(const v8h*)a1, up1 = *(const v8h*)(a1 + 16);
    v16h af0, af1;
#pragma unroll
    for (int j = 0; j < 8; ++j) {
      af0[j] = lo0[j]; af0[j + 8] = up0[j];
      af1[j] = lo1[j]; af1[j + 8] = up1[j];
    }
#pragma unroll
    for (int t = 0; t < 4; ++t) {
      const _Float16* bb =
          Wp + (((size_t)((n0 >> 4) + t) * kchunks + kc) * 32 + lane) * 16;
      v16h bf = *(const v16h*)bb;   // coalesced 1KB / wave, 32B aligned
      acc[0][t] = __builtin_amdgcn_wmma_f32_16x16x32_f16(
          false, af0, false, bf, (short)0, acc[0][t], false, false);
      acc[1][t] = __builtin_amdgcn_wmma_f32_16x16x32_f16(
          false, af1, false, bf, (short)0, acc[1][t], false, false);
    }
  }

#pragma unroll
  for (int u = 0; u < 2; ++u) {
#pragma unroll
    for (int t = 0; t < 4; ++t) {
      const int n = n0 + t * 16 + col;
      const float b = bias ? bias[n] : 0.0f;
#pragma unroll
      for (int r = 0; r < 8; ++r) {
        float v = acc[u][t][r] + b;
        if (do_relu) v = fmaxf(v, 0.0f);
        const size_t ofs = (size_t)(m0 + u * 16 + r + 8 * hi) * ldc + n;
        if (C)   C[ofs]   = v;
        if (C16) C16[ofs] = (_Float16)v;
      }
    }
  }
}

// ---------------------------------------------------------------------------
// Weight packing into fragment-contiguous order (f32 -> f16).
// transposed=0: W stored [N,K] row-major (PyTorch w);  =1: W stored [K,N].
// ---------------------------------------------------------------------------
__global__ void pack_b_kernel(const float* __restrict__ W, _Float16* __restrict__ out,
                              int K, int N, int transposed) {
  int i = blockIdx.x * blockDim.x + threadIdx.x;
  if (i >= N * K) return;
  const int kchunks = K >> 5;
  int j    = i & 15;
  int lane = (i >> 4) & 31;
  int kc   = (i >> 9) % kchunks;
  int nt   = i / (kchunks << 9);
  int n = nt * 16 + (lane & 15);
  int k = kc * 32 + (j >> 3) * 16 + (lane >> 4) * 8 + (j & 7);
  float v = transposed ? W[(size_t)k * N + n] : W[(size_t)n * K + k];
  out[i] = (_Float16)v;
}

__global__ void zero2_kernel(float* acc) {
  if (threadIdx.x == 0 && blockIdx.x == 0) { acc[0] = 0.0f; acc[1] = 0.0f; }
}

// ---------------------------------------------------------------------------
// Pointwise GRU gate update (PyTorch GRUCell), maintains f16 shadow of h.
// ---------------------------------------------------------------------------
__device__ __forceinline__ float sigm_(float x) { return 1.0f / (1.0f + __expf(-x)); }

__global__ void gru_update_kernel(const float* __restrict__ gi, const float* __restrict__ gh,
                                  const float* __restrict__ bih, const float* __restrict__ bhh,
                                  float* __restrict__ h, _Float16* __restrict__ h16)
{
  int tid = blockIdx.x * blockDim.x + threadIdx.x;
  if (tid >= B_ * H_) return;
  int b = tid / H_, j = tid % H_;
  const float* gib = gi + (size_t)b * (3 * H_);
  const float* ghb = gh + (size_t)b * (3 * H_);
  float ir = gib[j]          + bih[j],          hr = ghb[j]          + bhh[j];
  float iz = gib[H_ + j]     + bih[H_ + j],     hz = ghb[H_ + j]     + bhh[H_ + j];
  float in = gib[2 * H_ + j] + bih[2 * H_ + j], hn = ghb[2 * H_ + j] + bhh[2 * H_ + j];
  float r = sigm_(ir + hr);
  float z = sigm_(iz + hz);
  float n = tanhf(in + r * hn);
  float hv = (1.0f - z) * n + z * h[tid];
  h[tid]   = hv;
  h16[tid] = (_Float16)hv;
}

// Observation GRU update; reads/writes global h by unique scatter indices.
__global__ void grud_update_scatter_kernel(const float* __restrict__ gi, const float* __restrict__ gh,
                                           const float* __restrict__ bih, const float* __restrict__ bhh,
                                           const int* __restrict__ idx,
                                           float* __restrict__ h, _Float16* __restrict__ h16)
{
  int tid = blockIdx.x * blockDim.x + threadIdx.x;
  if (tid >= O_ * H_) return;
  int o = tid / H_, j = tid % H_;
  const size_t g = (size_t)idx[o] * H_ + j;
  const float* gib = gi + (size_t)o * (3 * H_);
  const float* ghb = gh + (size_t)o * (3 * H_);
  float ir = gib[j]          + bih[j],          hr = ghb[j]          + bhh[j];
  float iz = gib[H_ + j]     + bih[H_ + j],     hz = ghb[H_ + j]     + bhh[H_ + j];
  float in = gib[2 * H_ + j] + bih[2 * H_ + j], hn = ghb[2 * H_ + j] + bhh[2 * H_ + j];
  float r = sigm_(ir + hr);
  float z = sigm_(iz + hz);
  float n = tanhf(in + r * hn);
  float hv = (1.0f - z) * n + z * h[g];
  h[g]   = hv;
  h16[g] = (_Float16)hv;
}

__global__ void gather_h16_kernel(const _Float16* __restrict__ h16, const int* __restrict__ idx,
                                  _Float16* __restrict__ out)
{
  int tid = blockIdx.x * blockDim.x + threadIdx.x;
  if (tid >= O_ * H_) return;
  int o = tid / H_, j = tid % H_;
  out[tid] = h16[(size_t)idx[o] * H_ + j];
}

// ---------------------------------------------------------------------------
// Observation prep + log-lik loss; writes f16 GRU input gi_in16[O, D*P].
// ---------------------------------------------------------------------------
__global__ __launch_bounds__(256) void obs_prep_loss1_kernel(
    const float* __restrict__ X, const float* __restrict__ M, const int* __restrict__ idx,
    const float* __restrict__ p, const float* __restrict__ w_prep,
    const float* __restrict__ bias_prep, _Float16* __restrict__ gi_in16,
    float* __restrict__ l1_acc)
{
  int tid = blockIdx.x * 256 + threadIdx.x;    // O_*D_ threads
  int o = tid / D_, d = tid % D_;
  int io = idx[o];
  float x = X[tid];
  float m = M[tid];
  float mean = p[(size_t)io * (2 * D_) + d];
  float logv = p[(size_t)io * (2 * D_) + D_ + d];
  float sigma = __expf(0.5f * logv);
  float err = (x - mean) / sigma;
  float c1 = 0.5f * (err * err + logv + 2.0f * LOGC_) * m;

  __shared__ float red[256];
  red[threadIdx.x] = c1;
  __syncthreads();
  for (int s2 = 128; s2 > 0; s2 >>= 1) {
    if ((int)threadIdx.x < s2) red[threadIdx.x] += red[threadIdx.x + s2];
    __syncthreads();
  }
  if (threadIdx.x == 0) atomicAdd(l1_acc, red[0]);

  float st[4] = {x, mean, logv, err};
  const float* wp = w_prep + d * 4 * P_;
  const float* bp = bias_prep + d * P_;
  _Float16* outp = gi_in16 + (size_t)o * (D_ * P_) + d * P_;
#pragma unroll
  for (int pp = 0; pp < P_; ++pp) {
    float v = bp[pp];
#pragma unroll
    for (int k = 0; k < 4; ++k) v += st[k] * wp[k * P_ + pp];
    outp[pp] = (_Float16)(fmaxf(v, 0.0f) * m);
  }
}

// KL loss vs fixed obs noise, after the observation update
__global__ __launch_bounds__(256) void obs_loss2_kernel(
    const float* __restrict__ X, const float* __restrict__ M, const int* __restrict__ idx,
    const float* __restrict__ p, float* __restrict__ l2_acc)
{
  int tid = blockIdx.x * 256 + threadIdx.x;    // O_*D_
  int o = tid / D_, d = tid % D_;
  int io = idx[o];
  float x = X[tid];
  float m = M[tid];
  float m2  = p[(size_t)io * (2 * D_) + d];
  float lv2 = p[(size_t)io * (2 * D_) + D_ + d];
  float kl = logf(OBSSTD_) - 0.5f * lv2
           + (__expf(lv2) + (m2 - x) * (m2 - x)) / (2.0f * OBSSTD_ * OBSSTD_)
           - 0.5f;
  float c = kl * m;

  __shared__ float red[256];
  red[threadIdx.x] = c;
  __syncthreads();
  for (int s2 = 128; s2 > 0; s2 >>= 1) {
    if ((int)threadIdx.x < s2) red[threadIdx.x] += red[threadIdx.x + s2];
    __syncthreads();
  }
  if (threadIdx.x == 0) atomicAdd(l2_acc, red[0]);
}

// ---------------------------------------------------------------------------
// Small dense layers (covariate embedding / classifier) — plain kernels
// ---------------------------------------------------------------------------
__global__ void cov1_kernel(const float* __restrict__ cov, const float* __restrict__ W1,
                            const float* __restrict__ b1, float* __restrict__ t1) {
  int tid = blockIdx.x * blockDim.x + threadIdx.x;     // B_*COVH_
  if (tid >= B_ * COVH_) return;
  int b = tid / COVH_, j = tid % COVH_;
  float acc = b1[j];
  for (int c = 0; c < COV_; ++c) acc += cov[b * COV_ + c] * W1[c * COVH_ + j];
  t1[tid] = fmaxf(acc, 0.0f);
}
__global__ void cov2_kernel(const float* __restrict__ t1, const float* __restrict__ W2,
                            const float* __restrict__ b2,
                            float* __restrict__ h, _Float16* __restrict__ h16) {
  int tid = blockIdx.x * blockDim.x + threadIdx.x;     // B_*H_
  if (tid >= B_ * H_) return;
  int b = tid / H_, j = tid % H_;
  float acc = b2[j];
  for (int c = 0; c < COVH_; ++c) acc += t1[b * COVH_ + c] * W2[c * H_ + j];
  float hv = tanhf(acc);
  h[tid]   = hv;
  h16[tid] = (_Float16)hv;
}
__global__ void cls1_kernel(const float* __restrict__ h, const float* __restrict__ W1,
                            const float* __restrict__ b1, float* __restrict__ t) {
  int tid = blockIdx.x * blockDim.x + threadIdx.x;     // B_*CLSH_
  if (tid >= B_ * CLSH_) return;
  int b = tid / CLSH_, j = tid % CLSH_;
  float acc = b1[j];
  for (int c = 0; c < H_; ++c) acc += h[(size_t)b * H_ + c] * W1[c * CLSH_ + j];
  t[tid] = fmaxf(acc, 0.0f);
}
__global__ void cls2_kernel(const float* __restrict__ t, const float* __restrict__ W2,
                            const float* __restrict__ b2, float* __restrict__ out) {
  int b = blockIdx.x * blockDim.x + threadIdx.x;
  if (b >= B_) return;
  float acc = b2[0];
  for (int c = 0; c < CLSH_; ++c) acc += t[b * CLSH_ + c] * W2[c];
  out[b] = acc;
}

__global__ void copy_kernel(const float* __restrict__ src, float* __restrict__ dst, int n) {
  int i = blockIdx.x * blockDim.x + threadIdx.x;
  if (i < n) dst[i] = src[i];
}
__global__ void write_loss_kernel(const float* __restrict__ acc, float* __restrict__ dst) {
  if (threadIdx.x == 0 && blockIdx.x == 0) dst[0] = acc[0] + acc[1];
}

// ---------------------------------------------------------------------------
// Host driver
// ---------------------------------------------------------------------------
extern "C" void kernel_launch(void* const* d_in, const int* in_sizes, int n_in,
                              void* d_out, int out_size, void* d_ws, size_t ws_size,
                              hipStream_t stream) {
  (void)in_sizes; (void)n_in; (void)out_size; (void)ws_size;

  const float* X        = (const float*)d_in[0];
  const float* Mmask    = (const float*)d_in[1];
  const int*   obs_idx  = (const int*)  d_in[2];
  const float* cov      = (const float*)d_in[3];
  const float* cov_W1   = (const float*)d_in[4];
  const float* cov_b1   = (const float*)d_in[5];
  const float* cov_W2   = (const float*)d_in[6];
  const float* cov_b2   = (const float*)d_in[7];
  const float* p_W1     = (const float*)d_in[8];
  const float* p_b1     = (const float*)d_in[9];
  const float* p_W2     = (const float*)d_in[10];
  const float* p_b2     = (const float*)d_in[11];
  const float* cls_W1   = (const float*)d_in[12];
  const float* cls_b1   = (const float*)d_in[13];
  const float* cls_W2   = (const float*)d_in[14];
  const float* cls_b2   = (const float*)d_in[15];
  const float* gru_wih  = (const float*)d_in[16];
  const float* gru_whh  = (const float*)d_in[17];
  const float* gru_bih  = (const float*)d_in[18];
  const float* gru_bhh  = (const float*)d_in[19];
  const float* grud_wih = (const float*)d_in[20];
  const float* grud_whh = (const float*)d_in[21];
  const float* grud_bih = (const float*)d_in[22];
  const float* grud_bhh = (const float*)d_in[23];
  const float* w_prep   = (const float*)d_in[24];
  const float* bias_prep= (const float*)d_in[25];

  float* out = (float*)d_out;

  // ---- workspace layout (256B-aligned slices) ----
  char* ws = (char*)d_ws;
  size_t off = 0;
  auto alloc = [&](size_t bytes) -> void* {
    void* ptr = ws + off;
    off = (off + bytes + 255) & ~(size_t)255;
    return ptr;
  };
  _Float16* wp_gru_wih  = (_Float16*)alloc((size_t)3 * H_ * 2 * D_ * 2);
  _Float16* wp_gru_whh  = (_Float16*)alloc((size_t)3 * H_ * H_ * 2);
  _Float16* wp_grud_wih = (_Float16*)alloc((size_t)3 * H_ * D_ * P_ * 2);
  _Float16* wp_grud_whh = (_Float16*)alloc((size_t)3 * H_ * H_ * 2);
  _Float16* wp_pW1      = (_Float16*)alloc((size_t)PH_ * H_ * 2);
  _Float16* wp_pW2      = (_Float16*)alloc((size_t)2 * D_ * PH_ * 2);

  float*    h       = (float*)   alloc((size_t)B_ * H_ * 4);
  _Float16* h16     = (_Float16*)alloc((size_t)B_ * H_ * 2);
  float*    p       = (float*)   alloc((size_t)B_ * 2 * D_ * 4);
  _Float16* p16     = (_Float16*)alloc((size_t)B_ * 2 * D_ * 2);
  _Float16* t_ph16  = (_Float16*)alloc((size_t)B_ * PH_ * 2);
  float*    gi      = (float*)   alloc((size_t)B_ * 3 * H_ * 4);  // reused for obs
  float*    gh      = (float*)   alloc((size_t)B_ * 3 * H_ * 4);
  _Float16* gi_in16 = (_Float16*)alloc((size_t)O_ * D_ * P_ * 2);
  _Float16* h_g16   = (_Float16*)alloc((size_t)O_ * H_ * 2);
  float*    t_cov   = (float*)   alloc((size_t)B_ * COVH_ * 4);
  float*    t_cls   = (float*)   alloc((size_t)B_ * CLSH_ * 4);
  float*    lacc    = (float*)   alloc(2 * 4);                    // l1, l2

  auto gemm = [&](const _Float16* A, int lda, const _Float16* Wp, const float* bias,
                  float* C, _Float16* C16, int ldc, int M_, int N_, int K_, int relu) {
    dim3 grid(M_ / 256, N_ / 64);
    wmma_gemm_kernel<<<grid, 256, 0, stream>>>(A, lda, Wp, bias, C, C16, ldc, K_, relu);
  };
  auto p_model = [&](void) {
    gemm(h16, H_, wp_pW1, p_b1, nullptr, t_ph16, PH_, B_, PH_, H_, 1);
    gemm(t_ph16, PH_, wp_pW2, p_b2, p, p16, 2 * D_, B_, 2 * D_, PH_, 0);
  };

  // ---- per-call init: zero loss accumulators, pack weights ----
  zero2_kernel<<<1, 1, 0, stream>>>(lacc);

  auto pack = [&](const float* src, _Float16* dst, int K_, int N_, int transposed) {
    int n = K_ * N_;
    pack_b_kernel<<<(n + 255) / 256, 256, 0, stream>>>(src, dst, K_, N_, transposed);
  };
  pack(gru_wih,  wp_gru_wih,  2 * D_,   3 * H_, 0);
  pack(gru_whh,  wp_gru_whh,  H_,       3 * H_, 0);
  pack(grud_wih, wp_grud_wih, D_ * P_,  3 * H_, 0);
  pack(grud_whh, wp_grud_whh, H_,       3 * H_, 0);
  pack(p_W1,     wp_pW1,      H_,       PH_,    1);
  pack(p_W2,     wp_pW2,      PH_,      2 * D_, 1);

  // ---- h0 = tanh(relu(cov@W1+b1)@W2+b2);  p0 = p_model(h0) ----
  cov1_kernel<<<(B_ * COVH_ + 255) / 256, 256, 0, stream>>>(cov, cov_W1, cov_b1, t_cov);
  cov2_kernel<<<(B_ * H_ + 255) / 256, 256, 0, stream>>>(t_cov, cov_W2, cov_b2, h, h16);
  p_model();

  const int BH_blocks = (B_ * H_ + 255) / 256;
  const int OH_blocks = (O_ * H_ + 255) / 256;
  const int OD_blocks = O_ * D_ / 256;

  // ---- sequential scan over time steps ----
  for (int s = 0; s < S_; ++s) {
    const float* Xs = X + (size_t)s * O_ * D_;
    const float* Ms = Mmask + (size_t)s * O_ * D_;
    const int*   Is = obs_idx + (size_t)s * O_;

    // continuous GRU Euler step: h = GRUCell(p, h)
    gemm(p16, 2 * D_, wp_gru_wih, nullptr, gi, nullptr, 3 * H_, B_, 3 * H_, 2 * D_, 0);
    gemm(h16, H_,     wp_gru_whh, nullptr, gh, nullptr, 3 * H_, B_, 3 * H_, H_,     0);
    gru_update_kernel<<<BH_blocks, 256, 0, stream>>>(gi, gh, gru_bih, gru_bhh, h, h16);
    p_model();

    // observation prep + log-lik loss
    obs_prep_loss1_kernel<<<OD_blocks, 256, 0, stream>>>(Xs, Ms, Is, p, w_prep, bias_prep,
                                                         gi_in16, &lacc[0]);
    gather_h16_kernel<<<OH_blocks, 256, 0, stream>>>(h16, Is, h_g16);

    // observation GRU (reuse gi/gh buffers, M=1024 rows)
    gemm(gi_in16, D_ * P_, wp_grud_wih, nullptr, gi, nullptr, 3 * H_, O_, 3 * H_, D_ * P_, 0);
    gemm(h_g16,   H_,      wp_grud_whh, nullptr, gh, nullptr, 3 * H_, O_, 3 * H_, H_,      0);
    grud_update_scatter_kernel<<<OH_blocks, 256, 0, stream>>>(gi, gh, grud_bih, grud_bhh,
                                                              Is, h, h16);
    p_model();

    // KL loss
    obs_loss2_kernel<<<OD_blocks, 256, 0, stream>>>(Xs, Ms, Is, p, &lacc[1]);
  }

  // ---- classifier + outputs: d_out = [h (B*H), loss (1), class_pred (B)] ----
  cls1_kernel<<<(B_ * CLSH_ + 255) / 256, 256, 0, stream>>>(h, cls_W1, cls_b1, t_cls);
  cls2_kernel<<<(B_ + 255) / 256, 256, 0, stream>>>(t_cls, cls_W2, cls_b2,
                                                    out + (size_t)B_ * H_ + 1);
  copy_kernel<<<BH_blocks, 256, 0, stream>>>(h, out, B_ * H_);
  write_loss_kernel<<<1, 1, 0, stream>>>(lacc, out + (size_t)B_ * H_);
}
